// MeshEncoder_7679401525446
// MI455X (gfx1250) — compile-verified, matrix operand order checked by reading, not run on hardware
//
#include <hip/hip_runtime.h>
#include <hip/hip_bf16.h>

// ---------------------------------------------------------------------------
// GraphSAGE (mean) x5 + BN/ReLU for MI455X (gfx1250, wave32).
// GEMMs: bf16 WMMA 16x16x32 (f32 accum), double-buffered LDS staging via
// global_load_async_to_lds_b128 (ASYNCcnt). Activations/weights converted to
// bf16 once per layer (K padded to 32); mean-divide folded into conversion.
// ---------------------------------------------------------------------------

typedef __attribute__((ext_vector_type(16))) __bf16 v16bf;
typedef __attribute__((ext_vector_type(8)))  float  v8f;

union Frag16 {
    v16bf v;
    uint4 q[2];
};

// f32 -> bf16 round-to-nearest-even
__device__ __forceinline__ unsigned short f2bf(float f) {
    unsigned u = __builtin_bit_cast(unsigned, f);
    u += 0x7FFFu + ((u >> 16) & 1u);
    return (unsigned short)(u >> 16);
}

// ------------------------------- utility ----------------------------------

__global__ void k_zero(float* __restrict__ p, long long n) {
    long long i = (long long)blockIdx.x * blockDim.x + threadIdx.x;
    if (i < n) p[i] = 0.0f;
}

__global__ void k_count(const int* __restrict__ dst, float* __restrict__ cnt, int e) {
    int i = blockIdx.x * blockDim.x + threadIdx.x;
    if (i < e) atomicAdd(&cnt[dst[i]], 1.0f);
}

__global__ void k_invert(float* __restrict__ cnt, int n) {
    int i = blockIdx.x * blockDim.x + threadIdx.x;
    if (i < n) cnt[i] = 1.0f / fmaxf(cnt[i], 1.0f);
}

// ----------------------------- edge scatter --------------------------------
__global__ void k_scatter(const float* __restrict__ x,
                          const int* __restrict__ src, const int* __restrict__ dst,
                          float* __restrict__ agg, long long total, int fin) {
    long long t = (long long)blockIdx.x * blockDim.x + threadIdx.x;
    if (t >= total) return;
    int chunks = fin >> 2;
    int ed = (int)(t / chunks);
    int c  = (int)(t % chunks) * 4;
    int s = src[ed], d = dst[ed];
    float4 v = *(const float4*)(x + (size_t)s * fin + c);
    float* o = agg + (size_t)d * fin + c;
    atomicAdd(o + 0, v.x);
    atomicAdd(o + 1, v.y);
    atomicAdd(o + 2, v.z);
    atomicAdd(o + 3, v.w);
}

// ----------------------- f32 -> bf16 staging (pad K) -----------------------
// out[r][0..kp) = bf16(in[r][0..fin) * rowscale[r]), zero pad to kp (kp%32==0).
__global__ void k_cvt_bf16(const float* __restrict__ in, const float* __restrict__ rowscale,
                           unsigned short* __restrict__ out, int rows, int fin, int kp) {
    long long t = (long long)blockIdx.x * blockDim.x + threadIdx.x;
    int kq = kp >> 2;
    if (t >= (long long)rows * kq) return;
    int r = (int)(t / kq);
    int k = (int)(t % kq) * 4;
    float4 v = make_float4(0.f, 0.f, 0.f, 0.f);
    if (k < fin) v = *(const float4*)(in + (size_t)r * fin + k);
    float s = rowscale ? rowscale[r] : 1.0f;
    ushort4 o;
    o.x = f2bf(v.x * s); o.y = f2bf(v.y * s);
    o.z = f2bf(v.z * s); o.w = f2bf(v.w * s);
    *(ushort4*)(out + (size_t)r * kp + k) = o;
}

// ------------------------------- fused GEMM --------------------------------
// out[n,fout] = abf @ wl^T + xbf @ wr^T + bias        (all bf16, K padded)
// Block 256 thr = 8 waves; block tile 128(M) x 64(N); wave tile 32x32;
// K-step 32; double-buffered LDS filled by async global->LDS b128 loads.
#define TM 128
#define TN 64
#define LDP 40  // padded LDS row (80 B): b128-aligned, conflict-free frag reads

__global__ __launch_bounds__(256)
void k_sage_gemm_bf16(const unsigned short* __restrict__ abf,
                      const unsigned short* __restrict__ xbf,
                      const unsigned short* __restrict__ wl,
                      const unsigned short* __restrict__ wr,
                      const float* __restrict__ bias,
                      float* __restrict__ out, int n, int kp, int fout) {
    __shared__ __attribute__((aligned(16))) unsigned short As[2][TM * LDP];
    __shared__ __attribute__((aligned(16))) unsigned short Bs[2][TN * LDP];

    const int tid  = threadIdx.x;
    const int wave = tid >> 5;
    const int lane = tid & 31;
    const int bm = blockIdx.x * TM;
    const int bn = blockIdx.y * TN;
    const int wm = (wave >> 1) * 32;  // 0,32,64,96
    const int wn = (wave & 1) * 32;   // 0,32

    const int ktiles = kp >> 5;       // K-tiles per segment
    const int T = 2 * ktiles;         // seg0: agg x Wl, seg1: x x Wr

    // per-thread 16B-chunk coordinates for tile staging
    const int cc = (tid & 3) * 8;     // chunk col in bf16 elems (0,8,16,24)
    const int r0 = tid >> 2;          // 0..63

    // clamp A rows so async loads stay in-bounds (bad rows never stored)
    int ar0 = bm + r0;      ar0 = ar0 < n ? ar0 : n - 1;
    int ar1 = bm + r0 + 64; ar1 = ar1 < n ? ar1 : n - 1;
    const int br = bn + r0;           // fout % 64 == 0 -> always in range

    v8f acc[2][2];
#pragma unroll
    for (int a = 0; a < 2; ++a)
#pragma unroll
        for (int b = 0; b < 2; ++b)
#pragma unroll
            for (int r = 0; r < 8; ++r) acc[a][b][r] = 0.0f;

    const int lm = lane & 15;
    const int kh = lane >> 4;
    const int kb = kh * 8;

    // issue one tile's async loads (3 x b128 per lane -> ASYNCcnt += 3)
    auto issue_tile = [&](int t, int buf) {
        int seg = (t >= ktiles) ? 1 : 0;
        int k0  = (t - seg * ktiles) << 5;
        const unsigned short* Asrc = seg ? xbf : abf;
        const unsigned short* Bsrc = seg ? wr  : wl;

        unsigned long long ga0 =
            (unsigned long long)(size_t)(Asrc + (size_t)ar0 * kp + k0 + cc);
        unsigned la0 = (unsigned)(size_t)&As[buf][r0 * LDP + cc];
        asm volatile("global_load_async_to_lds_b128 %0, %1, off"
                     :: "v"(la0), "v"(ga0) : "memory");

        unsigned long long ga1 =
            (unsigned long long)(size_t)(Asrc + (size_t)ar1 * kp + k0 + cc);
        unsigned la1 = (unsigned)(size_t)&As[buf][(r0 + 64) * LDP + cc];
        asm volatile("global_load_async_to_lds_b128 %0, %1, off"
                     :: "v"(la1), "v"(ga1) : "memory");

        unsigned long long gb =
            (unsigned long long)(size_t)(Bsrc + (size_t)br * kp + k0 + cc);
        unsigned lb = (unsigned)(size_t)&Bs[buf][r0 * LDP + cc];
        asm volatile("global_load_async_to_lds_b128 %0, %1, off"
                     :: "v"(lb), "v"(gb) : "memory");
    };

    issue_tile(0, 0);

    for (int t = 0; t < T; ++t) {
        const int buf = t & 1;
        if (t + 1 < T) {
            issue_tile(t + 1, buf ^ 1);            // prefetch next tile
            asm volatile("s_wait_asynccnt 0x3" ::: "memory");  // tile t landed
        } else {
            asm volatile("s_wait_asynccnt 0x0" ::: "memory");
        }
        __syncthreads();

        Frag16 fa[2], fb[2];
#pragma unroll
        for (int q = 0; q < 2; ++q) {
            const unsigned short* ra = &As[buf][(wm + q * 16 + lm) * LDP];
            fa[q].q[0] = *(const uint4*)(ra + kb);
            fa[q].q[1] = *(const uint4*)(ra + 16 + kb);
            const unsigned short* rb = &Bs[buf][(wn + q * 16 + lm) * LDP];
            fb[q].q[0] = *(const uint4*)(rb + kb);
            fb[q].q[1] = *(const uint4*)(rb + 16 + kb);
        }
#pragma unroll
        for (int tm = 0; tm < 2; ++tm)
#pragma unroll
            for (int tn = 0; tn < 2; ++tn)
                acc[tm][tn] = __builtin_amdgcn_wmma_f32_16x16x32_bf16(
                    false, fa[tm].v, false, fb[tn].v,
                    (short)0, acc[tm][tn], false, false);
        __syncthreads();   // buffer `buf` free for prefetch at t+2
    }

    // epilogue: +bias, coalesced f32 stores across the 16 N-lanes
#pragma unroll
    for (int tn = 0; tn < 2; ++tn) {
        int gn = bn + wn + tn * 16 + lm;
        float bv = bias[gn];
#pragma unroll
        for (int tm = 0; tm < 2; ++tm) {
#pragma unroll
            for (int r = 0; r < 8; ++r) {
                int gm = bm + wm + tm * 16 + r + 8 * kh;
                if (gm < n) out[(size_t)gm * fout + gn] = acc[tm][tn][r] + bv;
            }
        }
    }
}

// ------------------------------ batch norm ---------------------------------

__global__ void k_bn_stats(const float* __restrict__ y, float* __restrict__ sum,
                           float* __restrict__ sumsq, int n, int f) {
    long long t = (long long)blockIdx.x * blockDim.x + threadIdx.x;
    int rowblocks = (n + 255) / 256;
    if (t >= (long long)f * rowblocks) return;
    int c  = (int)(t % f);
    int rb = (int)(t / f);
    int r0 = rb * 256;
    int r1 = (r0 + 256 < n) ? r0 + 256 : n;
    float s = 0.f, s2 = 0.f;
    for (int r = r0; r < r1; ++r) {
        float v = y[(size_t)r * f + c];
        s += v; s2 += v * v;
    }
    atomicAdd(&sum[c], s);
    atomicAdd(&sumsq[c], s2);
}

__global__ void k_bn_finalize(const float* __restrict__ sum, const float* __restrict__ sumsq,
                              const float* __restrict__ g, const float* __restrict__ be,
                              float* __restrict__ scalev, float* __restrict__ shiftv,
                              int n, int f) {
    int c = blockIdx.x * blockDim.x + threadIdx.x;
    if (c >= f) return;
    float inv_n = 1.0f / (float)n;
    float m = sum[c] * inv_n;
    float v = sumsq[c] * inv_n - m * m;
    float sc = g[c] * rsqrtf(v + 1e-5f);
    scalev[c] = sc;
    shiftv[c] = be[c] - m * sc;
}

__global__ void k_bn_apply_relu(float* __restrict__ y, const float* __restrict__ scalev,
                                const float* __restrict__ shiftv, long long total4, int f) {
    long long i = (long long)blockIdx.x * blockDim.x + threadIdx.x;
    if (i >= total4) return;
    int c = (int)((i * 4) % f);
    float4 v = ((float4*)y)[i];
    v.x = fmaxf(0.f, v.x * scalev[c + 0] + shiftv[c + 0]);
    v.y = fmaxf(0.f, v.y * scalev[c + 1] + shiftv[c + 1]);
    v.z = fmaxf(0.f, v.z * scalev[c + 2] + shiftv[c + 2]);
    v.w = fmaxf(0.f, v.w * scalev[c + 3] + shiftv[c + 3]);
    ((float4*)y)[i] = v;
}

// ------------------------------- launcher ----------------------------------

static inline int cdiv_ll(long long a, int b) { return (int)((a + b - 1) / b); }

extern "C" void kernel_launch(void* const* d_in, const int* in_sizes, int n_in,
                              void* d_out, int out_size, void* d_ws, size_t ws_size,
                              hipStream_t stream) {
    const int dims[6] = {196, 64, 128, 256, 256, 576};

    const float* x0 = (const float*)d_in[0];
    const int*   ei = (const int*)d_in[1];
    const int n = in_sizes[0] / dims[0];
    const int e = in_sizes[1] / 2;
    const int* src = ei;
    const int* dst = ei + e;

    const float *W[5][2], *B[5], *G[4], *BE[4];
    int idx = 2;
    for (int i = 0; i < 5; ++i) {
        W[i][0] = (const float*)d_in[idx++];
        W[i][1] = (const float*)d_in[idx++];
        B[i]    = (const float*)d_in[idx++];
    }
    for (int i = 0; i < 4; ++i) {
        G[i]  = (const float*)d_in[idx++];
        BE[i] = (const float*)d_in[idx++];
    }

    // ---- workspace layout ----
    float* fb     = (float*)d_ws;
    float* cnt    = fb;                      fb += n;           // inverse degree
    float* bufA   = fb;                      fb += (size_t)n * 256;
    float* bufB   = fb;                      fb += (size_t)n * 256;
    float* agg    = fb;                      fb += (size_t)n * 256;
    float* colsum = fb;                      fb += 1024;
    float* colsq  = fb;                      fb += 1024;
    float* scalev = fb;                      fb += 1024;
    float* shiftv = fb;                      fb += 1024;
    // bf16 region (16B aligned)
    size_t off = ((size_t)fb + 15) & ~(size_t)15;
    unsigned short* abf  = (unsigned short*)off;     // n x 256
    unsigned short* xbf  = abf + (size_t)n * 256;    // n x 256
    unsigned short* wlbf = xbf + (size_t)n * 256;    // 576 x 256
    unsigned short* wrbf = wlbf + (size_t)576 * 256;

    // inverse in-degree (edges fixed across layers -> once)
    k_zero<<<cdiv_ll(n, 256), 256, 0, stream>>>(cnt, n);
    k_count<<<cdiv_ll(e, 256), 256, 0, stream>>>(dst, cnt, e);
    k_invert<<<cdiv_ll(n, 256), 256, 0, stream>>>(cnt, n);

    const float* xin = x0;
    float* outs[5] = {bufA, bufB, bufA, bufB, (float*)d_out};

    for (int L = 0; L < 5; ++L) {
        const int fin = dims[L], fout = dims[L + 1];
        const int kp = (fin + 31) & ~31;     // padded K
        float* out = outs[L];

        long long na = (long long)n * fin;
        k_zero<<<cdiv_ll(na, 256), 256, 0, stream>>>(agg, na);

        long long tot = (long long)e * (fin >> 2);
        k_scatter<<<cdiv_ll(tot, 256), 256, 0, stream>>>(xin, src, dst, agg, tot, fin);

        // bf16 staging: agg*(1/deg), x, Wl, Wr
        long long ca = (long long)n * (kp >> 2);
        k_cvt_bf16<<<cdiv_ll(ca, 256), 256, 0, stream>>>(agg, cnt, abf, n, fin, kp);
        k_cvt_bf16<<<cdiv_ll(ca, 256), 256, 0, stream>>>(xin, nullptr, xbf, n, fin, kp);
        long long cw = (long long)fout * (kp >> 2);
        k_cvt_bf16<<<cdiv_ll(cw, 256), 256, 0, stream>>>(W[L][0], nullptr, wlbf, fout, fin, kp);
        k_cvt_bf16<<<cdiv_ll(cw, 256), 256, 0, stream>>>(W[L][1], nullptr, wrbf, fout, fin, kp);

        dim3 g(cdiv_ll(n, TM), fout / TN);
        k_sage_gemm_bf16<<<g, 256, 0, stream>>>(abf, xbf, wlbf, wrbf, B[L],
                                                out, n, kp, fout);

        if (L < 4) {
            k_zero<<<cdiv_ll(2048, 256), 256, 0, stream>>>(colsum, 2048); // colsum+colsq
            int rowblocks = (n + 255) / 256;
            long long st = (long long)fout * rowblocks;
            k_bn_stats<<<cdiv_ll(st, 256), 256, 0, stream>>>(out, colsum, colsq, n, fout);
            k_bn_finalize<<<cdiv_ll(fout, 256), 256, 0, stream>>>(colsum, colsq, G[L], BE[L],
                                                                  scalev, shiftv, n, fout);
            long long t4 = (long long)n * fout / 4;
            k_bn_apply_relu<<<cdiv_ll(t4, 256), 256, 0, stream>>>(out, scalev, shiftv, t4, fout);
        }
        xin = out;
    }
}
